// GetInitialSymbol_60593398612565
// MI455X (gfx1250) — matrix operand order repaired and checked by reading, not compiled, not run on hardware
//
#include <hip/hip_runtime.h>

#define DEV_INLINE __device__ __forceinline__

typedef __attribute__((ext_vector_type(16))) _Float16 v16h;
typedef __attribute__((ext_vector_type(8)))  _Float16 v8h;
typedef __attribute__((ext_vector_type(8)))  float    v8f;
typedef __attribute__((ext_vector_type(4)))  float    v4f;

constexpr int B_   = 8;
constexpr int T_   = 2048;
constexpr int S_   = 2048;
constexpr int D_   = 512;        // d_in
constexpr int HQ_  = 33;         // w_pos + 1 (real head dim)
constexpr int H_   = 64;         // head dim padded for WMMA K-steps
constexpr int SP1_ = S_ + 1;     // keys including the prepended (k0,v0) token
constexpr int SPAD_ = 2080;      // SP1 padded up to a multiple of 32
constexpr int NKB_  = SPAD_ / 32; // 65 key blocks of 32

// ---------------------------------------------------------------- helpers --
DEV_INLINE v8f wmma_f16(v16h a, v16h b, v8f c) {
  // D(f32 16x16) = A(f16 16x32) * B(f16 32x16) + C
  return __builtin_amdgcn_wmma_f32_16x16x32_f16(
      /*neg_a=*/false, a, /*neg_b=*/false, b,
      /*c_mod=*/(short)0, c, /*reuse_a=*/false, /*reuse_b=*/false);
}

// A-fragment gather for a 16x32 f16 tile held row-major in memory.
// lane L: row M = L&15, half = L>>4. a[i] -> kdim = half*16 + i,
// so chunk0 = row[off .. off+7], chunk1 = row[off+8 .. off+15], off = half*16 (+32*kstep).
DEV_INLINE v16h load_afrag(const _Float16* row, int off) {
  v8h c0 = *(const v8h*)(row + off);
  v8h c1 = *(const v8h*)(row + off + 8);
  v16h a;
#pragma unroll
  for (int i = 0; i < 8; ++i) { a[i] = c0[i]; a[8 + i] = c1[i]; }
  return a;
}

// ------------------------------------------------------------ projections --
// Qh[b][t][h] = f16( tgt[b][t] . w_q[h] + b_q[h] ), zero for h >= 33.
__global__ __launch_bounds__(64) void qproj_kernel(
    const float* __restrict__ tgt, const float* __restrict__ w_q,
    const float* __restrict__ b_q, _Float16* __restrict__ Qh) {
  const int t = blockIdx.x, b = blockIdx.y, h = threadIdx.x;
  float sum = 0.0f;
  if (h < HQ_) {
    sum = b_q[h];
    const v4f* tr = (const v4f*)(tgt + ((size_t)(b * T_ + t)) * D_);
    const v4f* wr = (const v4f*)(w_q + (size_t)h * D_);
#pragma unroll 4
    for (int c = 0; c < D_ / 4; ++c) {
      v4f a = tr[c], w = wr[c];
      sum += a[0] * w[0] + a[1] * w[1] + a[2] * w[2] + a[3] * w[3];
    }
  }
  Qh[((size_t)(b * T_ + t)) * H_ + h] = (_Float16)sum;
}

// Kh[b][s][h]: s==0 -> k_0; 1<=s<=S -> memory[s-1].w_k[h]+b_k[h]; else 0. h>=33 -> 0.
__global__ __launch_bounds__(64) void kproj_kernel(
    const float* __restrict__ memory, const float* __restrict__ w_k,
    const float* __restrict__ b_k, const float* __restrict__ k_0,
    _Float16* __restrict__ Kh) {
  const int s = blockIdx.x, b = blockIdx.y, h = threadIdx.x;
  float sum = 0.0f;
  if (s == 0) {
    if (h < HQ_) sum = k_0[h];
  } else if (s <= S_ && h < HQ_) {
    sum = b_k[h];
    const v4f* mr = (const v4f*)(memory + ((size_t)(b * S_ + (s - 1))) * D_);
    const v4f* wr = (const v4f*)(w_k + (size_t)h * D_);
#pragma unroll 4
    for (int c = 0; c < D_ / 4; ++c) {
      v4f a = mr[c], w = wr[c];
      sum += a[0] * w[0] + a[1] * w[1] + a[2] * w[2] + a[3] * w[3];
    }
  }
  Kh[((size_t)(b * SPAD_ + s)) * H_ + h] = (_Float16)sum;
}

// Vt[b][d][s] = f16( V[s][d] ), V[s] = s==0 ? v_0 : (s<=S ? memory[s-1]@w_v.T + b_v : 0)
// Computed transposed via WMMA: D(d x s) = w_v(d x c) * memory^T(c x s).
__global__ __launch_bounds__(32) void vproj_kernel(
    const float* __restrict__ memory, const float* __restrict__ w_v,
    const float* __restrict__ b_v, const float* __restrict__ v_0,
    _Float16* __restrict__ Vt) {
  const int s0 = blockIdx.x * 16, d0 = blockIdx.y * 16, b = blockIdx.z;
  const int lane = threadIdx.x & 31;
  const int col  = lane & 15;   // N column of tile = s index ; also A row index
  const int half = lane >> 4;

  const int s = s0 + col;
  const bool valid = (s >= 1 && s <= S_);
  const float* src  = memory + ((size_t)(b * S_ + (valid ? (s - 1) : 0))) * D_;
  const float* wrow = w_v + (size_t)(d0 + col) * D_;

  v8f acc;
#pragma unroll
  for (int r = 0; r < 8; ++r) acc[r] = 0.0f;

  for (int cb = 0; cb < D_; cb += 32) {
    // A = w_v fragment (M=d, K=c)
    v8f w0 = *(const v8f*)(wrow + cb + half * 16);
    v8f w1 = *(const v8f*)(wrow + cb + half * 16 + 8);
    v16h a;
#pragma unroll
    for (int i = 0; i < 8; ++i) { a[i] = (_Float16)w0[i]; a[8 + i] = (_Float16)w1[i]; }
    // B = memory^T fragment (K=c, N=s): b[i] = mem[s-1][cb + half*16 + i]
    v16h bf;
    if (valid) {
      const v4f* sp = (const v4f*)(src + cb + half * 16);
#pragma unroll
      for (int k2 = 0; k2 < 4; ++k2) {
        v4f t = sp[k2];
#pragma unroll
        for (int e = 0; e < 4; ++e) bf[4 * k2 + e] = (_Float16)t[e];
      }
    } else {
#pragma unroll
      for (int i = 0; i < 16; ++i) bf[i] = (_Float16)0.0f;
    }
    acc = wmma_f16(a, bf, acc);
  }

#pragma unroll
  for (int r = 0; r < 8; ++r) {
    const int d = d0 + half * 8 + r;
    float val;
    if (s == 0)           val = v_0[d];
    else if (valid)       val = acc[r] + b_v[d];
    else                  val = 0.0f;   // padded keys: exact zeros (no NaN into WMMA)
    Vt[((size_t)(b * D_ + d)) * SPAD_ + s] = (_Float16)val;
  }
}

// ---------------------------------------------------------- flash attention --
// WG = 16 queries, 4 waves; wave w owns output dims [w*128, w*128+128).
// Scores computed transposed: S^T(key x query) = K_tile(16x64) * Q^T(64x16),
// then out^T(d x query) += V^T_tile(16x32) * P^T(32x16).
__global__ __launch_bounds__(128) void attn_kernel(
    const float* __restrict__ tgt, const _Float16* __restrict__ Qh,
    const _Float16* __restrict__ Kh, const _Float16* __restrict__ Vt,
    float* __restrict__ out) {
  const int b    = blockIdx.y;
  const int t0   = blockIdx.x * 16;
  const int wave = threadIdx.x >> 5;
  const int lane = threadIdx.x & 31;
  const int col  = lane & 15;   // query column n (and A-matrix row m)
  const int half = lane >> 4;

  // Q^T B-fragments, loaded once: b[i] = Q[t0+col][h = half*16 + i (+32)]
  const _Float16* qrow = Qh + ((size_t)(b * T_ + t0 + col)) * H_;
  const v16h bq0 = *(const v16h*)(qrow + half * 16);
  const v16h bq1 = *(const v16h*)(qrow + 32 + half * 16);

  const _Float16* kbase = Kh + ((size_t)b * SPAD_ + col) * H_;            // key row s=col
  const _Float16* vbase = Vt + ((size_t)(b * D_ + wave * 128 + col)) * SPAD_;

  v8f acc[8];
#pragma unroll
  for (int j = 0; j < 8; ++j)
#pragma unroll
    for (int r = 0; r < 8; ++r) acc[j][r] = 0.0f;

  float m = -1e30f, l = 0.0f;

  for (int kb = 0; kb < NKB_; ++kb) {
    const int sblk = kb * 32;
    const _Float16* krow0 = kbase + (size_t)sblk * H_;
    const _Float16* krow1 = krow0 + (size_t)16 * H_;
    if (kb + 1 < NKB_) {
      __builtin_prefetch(krow0 + (size_t)32 * H_, 0, 1);                 // next K block
      __builtin_prefetch(vbase + (size_t)sblk + 32, 0, 1);               // next V chunk
    }

    // ---- S^T tiles: st0 = keys [sblk, sblk+16), st1 = keys [sblk+16, sblk+32)
    v8f st0, st1;
#pragma unroll
    for (int r = 0; r < 8; ++r) { st0[r] = 0.0f; st1[r] = 0.0f; }
    st0 = wmma_f16(load_afrag(krow0, half * 16), bq0, st0);
    st0 = wmma_f16(load_afrag(krow0, 32 + half * 16), bq1, st0);
    st1 = wmma_f16(load_afrag(krow1, half * 16), bq0, st1);
    st1 = wmma_f16(load_afrag(krow1, 32 + half * 16), bq1, st1);

    // ---- mask padded keys (s >= SP1). C-layout: reg r -> key = sblk + half*8 + r.
#pragma unroll
    for (int r = 0; r < 8; ++r) {
      if (sblk + half * 8 + r >= SP1_)      st0[r] = -1e30f;
      if (sblk + 16 + half * 8 + r >= SP1_) st1[r] = -1e30f;
    }

    // ---- online softmax (per query col; other 16 keys live in lane^16)
    float bmax = st0[0];
#pragma unroll
    for (int r = 1; r < 8; ++r) bmax = fmaxf(bmax, st0[r]);
#pragma unroll
    for (int r = 0; r < 8; ++r) bmax = fmaxf(bmax, st1[r]);
    bmax = fmaxf(bmax, __shfl_xor(bmax, 16, 32));
    const float mn   = fmaxf(m, bmax);
    const float corr = __expf(m - mn);
    m = mn;

    float p0[8], p1[8], bsum = 0.0f;
#pragma unroll
    for (int r = 0; r < 8; ++r) { p0[r] = __expf(st0[r] - mn); bsum += p0[r]; }
#pragma unroll
    for (int r = 0; r < 8; ++r) { p1[r] = __expf(st1[r] - mn); bsum += p1[r]; }
    bsum += __shfl_xor(bsum, 16, 32);
    l = l * corr + bsum;

#pragma unroll
    for (int j = 0; j < 8; ++j)
#pragma unroll
      for (int r = 0; r < 8; ++r) acc[j][r] *= corr;

    // ---- P^T B-fragment (K=32 keys, N=16 queries): one xor-16 shuffle per reg.
    v16h bp;
#pragma unroll
    for (int r = 0; r < 8; ++r) {
      const float o0 = __shfl_xor(p0[r], 16, 32);
      const float o1 = __shfl_xor(p1[r], 16, 32);
      bp[r]     = (_Float16)(half ? o1    : p0[r]);
      bp[8 + r] = (_Float16)(half ? p1[r] : o0);
    }

    // ---- out^T += V^T * P^T over this wave's 8 d-tiles
#pragma unroll
    for (int j = 0; j < 8; ++j) {
      const _Float16* vrow = vbase + (size_t)j * 16 * SPAD_ + sblk;
      acc[j] = wmma_f16(load_afrag(vrow, half * 16), bp, acc[j]);
    }
  }

  // ---- epilogue: out = tgt + acc / l
  const float rl = 1.0f / l;
  const int t = t0 + col;
  const float* trow = tgt + ((size_t)(b * T_ + t)) * D_;
  float*       orow = out + ((size_t)(b * T_ + t)) * D_;
#pragma unroll
  for (int j = 0; j < 8; ++j) {
    const int d = wave * 128 + j * 16 + half * 8;
    const v4f* tr = (const v4f*)(trow + d);
    v4f* op = (v4f*)(orow + d);
    v4f x0 = tr[0], x1 = tr[1], y0, y1;
#pragma unroll
    for (int e = 0; e < 4; ++e) {
      y0[e] = x0[e] + acc[j][e] * rl;
      y1[e] = x1[e] + acc[j][4 + e] * rl;
    }
    op[0] = y0;
    op[1] = y1;
  }
}

// ---------------------------------------------------------------- launcher --
extern "C" void kernel_launch(void* const* d_in, const int* in_sizes, int n_in,
                              void* d_out, int out_size, void* d_ws, size_t ws_size,
                              hipStream_t stream) {
  (void)in_sizes; (void)n_in; (void)out_size; (void)ws_size;
  const float* tgt    = (const float*)d_in[0];
  const float* memory = (const float*)d_in[1];
  const float* w_q    = (const float*)d_in[2];
  const float* b_q    = (const float*)d_in[3];
  const float* w_k    = (const float*)d_in[4];
  const float* b_k    = (const float*)d_in[5];
  const float* w_v    = (const float*)d_in[6];
  const float* b_v    = (const float*)d_in[7];
  const float* k_0    = (const float*)d_in[8];
  const float* v_0    = (const float*)d_in[9];
  float* out = (float*)d_out;

  // workspace (f16): Qh [B,T,64] | Kh [B,SPAD,64] | Vt [B,512,SPAD]  (~21.3 MB)
  _Float16* Qh = (_Float16*)d_ws;
  _Float16* Kh = Qh + (size_t)B_ * T_ * H_;
  _Float16* Vt = Kh + (size_t)B_ * SPAD_ * H_;

  qproj_kernel<<<dim3(T_, B_), 64, 0, stream>>>(tgt, w_q, b_q, Qh);
  kproj_kernel<<<dim3(SPAD_, B_), 64, 0, stream>>>(memory, w_k, b_k, k_0, Kh);
  vproj_kernel<<<dim3(SPAD_ / 16, D_ / 16, B_), 32, 0, stream>>>(memory, w_v, b_v, v_0, Vt);
  attn_kernel<<<dim3(T_ / 16, B_), 128, 0, stream>>>(tgt, Qh, Kh, Vt, out);
}
